// CGRNN_batch_adjIGraph_33741263078247
// MI455X (gfx1250) — matrix operand order repaired and checked by reading, not compiled, not run on hardware
//
#include <hip/hip_runtime.h>
#include <hip/hip_bf16.h>
#include <math.h>

// Shapes: B=32, T=64, N=128, D=64, E=5, K=2D+1=129 (padded to KP=160)
#define BATCH 32
#define TT 64
#define NN 128
#define DD 64
#define EE 5
#define KP 160   // padded K for gate matmuls (5 chunks of 32)

typedef __attribute__((ext_vector_type(16))) __bf16 v16bf;
typedef __attribute__((ext_vector_type(8)))  float  v8f;

__device__ __forceinline__ v8f wmma_bf16(v16bf a, v16bf b, v8f c) {
  return __builtin_amdgcn_wmma_f32_16x16x32_bf16(false, a, false, b, (short)0, c,
                                                 false, false);
}

// ---------------- Precompute kernels (time-invariant quantities) -------------

// vv[n,e] = relu(prior[n] @ W1 + b1) @ W2 + b2 ; beff[g][n][o] = sum_e vv[n,e]*b_g[e][o]
__global__ __launch_bounds__(128)
void pre_vv_kernel(const float* __restrict__ prior, const float* __restrict__ W1,
                   const float* __restrict__ b1, const float* __restrict__ W2,
                   const float* __restrict__ b2, const float* __restrict__ br_,
                   const float* __restrict__ bu_, const float* __restrict__ bc_,
                   float* __restrict__ vv, float* __restrict__ beff) {
  __shared__ float hid[128];
  __shared__ float vvs[8];
  int n = blockIdx.x, j = threadIdx.x;
  float acc = b1[j];
  for (int k = 0; k < 768; ++k) acc += prior[n * 768 + k] * W1[k * 128 + j];
  hid[j] = fmaxf(acc, 0.f);
  __syncthreads();
  if (j < EE) {
    float s = b2[j];
    for (int jj = 0; jj < 128; ++jj) s += hid[jj] * W2[jj * EE + j];
    vv[n * EE + j] = s;
    vvs[j] = s;
  }
  __syncthreads();
  if (j < DD) {
    float sr = 0.f, su = 0.f, sc = 0.f;
    for (int e = 0; e < EE; ++e) {
      float v = vvs[e];
      sr += v * br_[e * DD + j];
      su += v * bu_[e * DD + j];
      sc += v * bc_[e * DD + j];
    }
    beff[(0 * NN + n) * DD + j] = sr;
    beff[(1 * NN + n) * DD + j] = su;
    beff[(2 * NN + n) * DD + j] = sc;
  }
}

// WT[g][e][o][k] = W_g[e][k][o] as bf16, zero-padded to k<KP. g: 0=r,1=u,2=c
__global__ __launch_bounds__(256)
void pre_wt_kernel(const float* __restrict__ Wr_, const float* __restrict__ Wu_,
                   const float* __restrict__ Wc_, __bf16* __restrict__ WT) {
  int idx = blockIdx.x * 256 + threadIdx.x;
  const int total = 3 * EE * DD * KP;
  if (idx >= total) return;
  int k = idx % KP;
  int o = (idx / KP) % DD;
  int e = (idx / (KP * DD)) % EE;
  int g = idx / (KP * DD * EE);
  const float* W = (g == 0) ? Wr_ : ((g == 1) ? Wu_ : Wc_);
  float v = (k < 129) ? W[(e * 129 + k) * DD + o] : 0.f;
  WT[idx] = (__bf16)v;
}

// tot[b][n] = sum_t mask[b][t][n]
__global__ __launch_bounds__(128)
void pre_tot_kernel(const float* __restrict__ mask, float* __restrict__ tot) {
  int b = blockIdx.x, n = threadIdx.x;
  float s = 0.f;
  for (int t = 0; t < TT; ++t) s += mask[(b * TT + t) * NN + n];
  tot[b * NN + n] = s;
}

// ---------------- Main recurrent kernel: one workgroup per batch element ----
// 8 waves. Wave w owns output node rows [16w, 16w+16) for every matmul, so all
// comb/h/u/h1 data flow between phases 1-2-3 is intra-wave (in-order LDS) and
// only TWO block barriers are needed per timestep. Gate loops keep Ot OUTERMOST
// so only one accumulator pair is live at a time (peak VGPRs < 256, no spills).

__global__ __launch_bounds__(256)
void cgrnn_main_kernel(const float* __restrict__ obs, const float* __restrict__ mask,
                       const int* __restrict__ lengths, const float* __restrict__ avg,
                       const float* __restrict__ rarW, const float* __restrict__ adjIg,
                       const float* __restrict__ vv, const float* __restrict__ beff,
                       const float* __restrict__ tot, const __bf16* __restrict__ WT,
                       float* __restrict__ out) {
  extern __shared__ char smem_raw[];
  float*  sh_h    = (float*)smem_raw;            // [128][64]  recurrent state (fp32)
  float*  sh_u    = sh_h + NN * DD;              // [128][64]  u gate carry 2->3
  float*  sh_comb = sh_u + NN * DD;              // [128][160] comb (fp32, node-major)
  float*  sh_fc   = sh_comb + NN * KP;           // [128][160] [x,h1] (fp32, node-major)
  __bf16* sh_ft   = (__bf16*)(sh_fc + NN * KP);  // [160][128] [x,h] feature-major (B op)
  float*  sh_rs   = (float*)(sh_ft + KP * NN);   // [128]
  float*  sh_m    = sh_rs + NN;                  // [128]
  float*  sh_vv   = sh_m + NN;                   // [128][5]

  const int b    = blockIdx.x;
  const int tid  = threadIdx.x;
  const int lane = tid & 31;
  const int w    = tid >> 5;     // wave 0..7
  const int hh   = lane >> 4;    // half (0/1)
  const int l16  = lane & 15;
  const int len  = lengths[b];

  for (int i = tid; i < NN * DD; i += 256) sh_h[i] = 0.f;
  for (int i = tid; i < NN * EE; i += 256) sh_vv[i] = vv[i];
  __syncthreads();

  const float* totB = tot + b * NN;

  for (int t = 0; t < TT; ++t) {
    const float* obs_t = obs + (size_t)(b * TT + t) * NN * DD;
    const float* avg_t = avg + (size_t)(b * TT + t) * NN;
    const float* msk_t = mask + (size_t)(b * TT + t) * NN;

    // ---- phase 0 (single barrier): rs, m, featT (bf16 B-operand, feature-
    //      major), featc static columns. rs recomputed inline where needed.
    if (tid < NN) {
      int nn = tid;
      sh_rs[nn] = 0.5f * tanhf(avg_t[nn] / (totB[nn] + 1.f));
      sh_m[nn]  = msk_t[nn];
    }
    for (int i = tid; i < KP * NN; i += 256) {
      int f = i >> 7, j = i & 127;
      float v;
      if (f < DD)        v = obs_t[j * DD + f];
      else if (f == DD)  v = 0.5f * tanhf(avg_t[j] / (totB[j] + 1.f));
      else if (f < 129)  v = sh_h[j * DD + (f - DD - 1)];
      else               v = 0.f;
      sh_ft[f * NN + j] = (__bf16)v;
    }
    for (int i = tid; i < NN * KP; i += 256) {
      int nn = i / KP, k = i - nn * KP;
      if (k < DD)        sh_fc[i] = obs_t[nn * DD + k];
      else if (k == DD)  sh_fc[i] = 0.5f * tanhf(avg_t[nn] / (totB[nn] + 1.f));
      else if (k >= 129) sh_fc[i] = 0.f;
      // cols 65..128 (h1) are written in phase 2 every step
    }
    // warm L2 / WGP$ with next timestep's obs tile (global_prefetch_b8)
    if (t + 1 < TT) {
      const char* nxt = (const char*)(obs + (size_t)(b * TT + t + 1) * NN * DD);
      __builtin_prefetch(nxt + (size_t)tid * 128, 0, 1);
    }
    __syncthreads();

    // ---- phase 1: comb = adj @ [x,h]. Adjacency rows built directly in
    //      registers (each row consumed only by its owner wave).
    {
      const int arow = w * 16 + l16;
      const float rsi = sh_rs[arow];
      const float mi  = sh_m[arow];
      const float* adjRow = adjIg + arow * NN;
      const float* rarRow = rarW + arow * NN;
      v16bf Aadj[4];
#pragma unroll
      for (int kc = 0; kc < 4; ++kc) {
        const int j0 = kc * 32 + 16 * hh;  // A-layout: elems 0..7 -> K=16*half+q
        const int j1 = j0 + 8;             //           elems 8..15 -> K=16*half+8+q
#pragma unroll
        for (int q = 0; q < 8; ++q) {
          int ja = j0 + q;
          float va = (ja == arow)
                         ? 1.f
                         : adjRow[ja] * (1.f - rarRow[ja] * fabsf(rsi - sh_rs[ja])) *
                               mi * sh_m[ja];
          Aadj[kc][q] = (__bf16)va;
          int jb = j1 + q;
          float vb = (jb == arow)
                         ? 1.f
                         : adjRow[jb] * (1.f - rarRow[jb] * fabsf(rsi - sh_rs[jb])) *
                               mi * sh_m[jb];
          Aadj[kc][8 + q] = (__bf16)vb;
        }
      }
      for (int Ft = 0; Ft < 10; ++Ft) {
        v8f acc = {};
        const int fcol = Ft * 16 + l16;
#pragma unroll
        for (int kc = 0; kc < 4; ++kc) {
          v16bf Bv = *(const v16bf*)&sh_ft[fcol * NN + kc * 32 + 16 * hh];
          acc = wmma_bf16(Aadj[kc], Bv, acc);
        }
#pragma unroll
        for (int r = 0; r < 8; ++r)
          sh_comb[(w * 16 + 8 * hh + r) * KP + fcol] = acc[r];
      }
    }
    // no block barrier: phases 2/3 read only this wave's comb rows (in-order LDS)

    const int n = w * 16 + l16;  // A-matrix row for this lane
    float vvl[EE];
#pragma unroll
    for (int e = 0; e < EE; ++e) vvl[e] = sh_vv[n * EE + e];

    // ---- phase 2: r,u = sigmoid(sum_{e,k}(vv[n,e]*comb[n,k]) W_{r,u}[e][k,o]+b)
    for (int Ot = 0; Ot < 4; ++Ot) {
      const int o = Ot * 16 + l16;
      v8f accR = {}, accU = {};
      for (int e = 0; e < EE; ++e) {
        const float s = vvl[e];
        for (int kc = 0; kc < 5; ++kc) {
          const float* pa = &sh_comb[n * KP + kc * 32 + 16 * hh];
          v16bf A;
#pragma unroll
          for (int q = 0; q < 8; ++q) {
            A[q]     = (__bf16)(pa[q] * s);
            A[8 + q] = (__bf16)(pa[8 + q] * s);
          }
          const int wo = kc * 32 + 16 * hh;
          v16bf Br = *(const v16bf*)&WT[((0 * EE + e) * DD + o) * KP + wo];
          v16bf Bu = *(const v16bf*)&WT[((1 * EE + e) * DD + o) * KP + wo];
          accR = wmma_bf16(A, Br, accR);
          accU = wmma_bf16(A, Bu, accU);
        }
      }
#pragma unroll
      for (int r = 0; r < 8; ++r) {
        const int n2 = w * 16 + 8 * hh + r;
        float rg = accR[r] + beff[(0 * NN + n2) * DD + o];
        float ug = accU[r] + beff[(1 * NN + n2) * DD + o];
        rg = 1.f / (1.f + expf(-rg));
        ug = 1.f / (1.f + expf(-ug));
        const float hv = sh_h[n2 * DD + o];
        const bool  mm = sh_m[n2] > 0.f;
        const float h1 = mm ? rg * hv : hv;
        sh_h[n2 * DD + o] = h1;            // h <- h1 (element owned by this lane)
        sh_u[n2 * DD + o] = ug;
        sh_fc[n2 * KP + DD + 1 + o] = h1;  // [x,h1] columns 65..128
      }
    }
    // no block barrier: phase 3 reads only this wave's sh_fc/sh_h/sh_u rows

    // ---- phase 3: cand = tanh(gate_c([x,h1])); h <- (1-u)h1 + u*cand ------
    for (int Ot = 0; Ot < 4; ++Ot) {
      const int o = Ot * 16 + l16;
      v8f accC = {};
      for (int e = 0; e < EE; ++e) {
        const float s = vvl[e];
        for (int kc = 0; kc < 5; ++kc) {
          const float* pa = &sh_fc[n * KP + kc * 32 + 16 * hh];
          v16bf A;
#pragma unroll
          for (int q = 0; q < 8; ++q) {
            A[q]     = (__bf16)(pa[q] * s);
            A[8 + q] = (__bf16)(pa[8 + q] * s);
          }
          v16bf Bc =
              *(const v16bf*)&WT[((2 * EE + e) * DD + o) * KP + kc * 32 + 16 * hh];
          accC = wmma_bf16(A, Bc, accC);
        }
      }
#pragma unroll
      for (int r = 0; r < 8; ++r) {
        const int n2 = w * 16 + 8 * hh + r;
        const float cnd = tanhf(accC[r] + beff[(2 * NN + n2) * DD + o]);
        const float h1  = sh_h[n2 * DD + o];
        const float ug  = sh_u[n2 * DD + o];
        const bool  mm  = sh_m[n2] > 0.f;
        const float h2  = mm ? (1.f - ug) * h1 + ug * cnd : h1;
        sh_h[n2 * DD + o] = h2;
        if (t == len - 1) out[((size_t)b * NN + n2) * DD + o] = h2;
      }
    }
    __syncthreads();  // next step's phase 0 reads sh_h / rewrites rs,m,featT
  }
}

// ---------------- Host launcher ---------------------------------------------

extern "C" void kernel_launch(void* const* d_in, const int* in_sizes, int n_in,
                              void* d_out, int out_size, void* d_ws, size_t ws_size,
                              hipStream_t stream) {
  const float* obs    = (const float*)d_in[0];
  const float* mask   = (const float*)d_in[2];
  const int*   lens   = (const int*)d_in[5];
  const float* avg    = (const float*)d_in[6];
  const float* prior  = (const float*)d_in[7];
  const float* rarW   = (const float*)d_in[8];
  const float* adjIg  = (const float*)d_in[9];
  const float* p2sW1  = (const float*)d_in[10];
  const float* p2sb1  = (const float*)d_in[11];
  const float* p2sW2  = (const float*)d_in[12];
  const float* p2sb2  = (const float*)d_in[13];
  const float* Wu     = (const float*)d_in[14];
  const float* bu     = (const float*)d_in[15];
  const float* Wr     = (const float*)d_in[16];
  const float* br     = (const float*)d_in[17];
  const float* Wc     = (const float*)d_in[18];
  const float* bc     = (const float*)d_in[19];
  float* outp = (float*)d_out;

  // workspace carve: vv[640] f32 | beff[3*128*64] f32 | tot[32*128] f32 | WT bf16
  float*  vv   = (float*)d_ws;
  float*  beff = vv + NN * EE;
  float*  tot  = beff + 3 * NN * DD;
  __bf16* WT   = (__bf16*)(tot + BATCH * NN);
  // total ws use = 424,448 bytes

  pre_vv_kernel<<<NN, 128, 0, stream>>>(prior, p2sW1, p2sb1, p2sW2, p2sb2,
                                        br, bu, bc, vv, beff);
  const int wt_total = 3 * EE * DD * KP;
  pre_wt_kernel<<<(wt_total + 255) / 256, 256, 0, stream>>>(Wr, Wu, Wc, WT);
  pre_tot_kernel<<<BATCH, 128, 0, stream>>>(mask, tot);

  const size_t smem = (size_t)(NN * DD * 2 + NN * KP * 2) * 4   // h, u, comb, featc
                    + (size_t)(KP * NN) * 2                     // featT (bf16)
                    + (size_t)(NN + NN + NN * EE) * 4;          // rs, m, vv
  static_assert(((NN * DD * 2 + NN * KP * 2) * 4 + (KP * NN) * 2 +
                 (NN + NN + NN * EE) * 4) <= 320 * 1024, "LDS budget");
  (void)hipFuncSetAttribute((const void*)cgrnn_main_kernel,
                            hipFuncAttributeMaxDynamicSharedMemorySize, (int)smem);
  cgrnn_main_kernel<<<BATCH, 256, smem, stream>>>(obs, mask, lens, avg, rarW, adjIg,
                                                  vv, beff, tot, WT, outp);
}
